// LSTM_Model_76046690943278
// MI455X (gfx1250) — compile-verified, matrix operand order checked by reading, not run on hardware
//
#include <hip/hip_runtime.h>

// ---------------- problem constants ----------------
constexpr int BATCH = 64;
constexpr int SEQ   = 256;
constexpr int EMB   = 300;
constexpr int KX    = 320;          // EMB padded to multiple of 32 for bf16 WMMA
constexpr int HID   = 256;
constexpr int G4    = 1024;         // 4*HID
constexpr float FORGET_BIAS = 1.0f;

// ---------------- vector types for WMMA ----------------
typedef __bf16 v16bf __attribute__((ext_vector_type(16)));
typedef float  v8f   __attribute__((ext_vector_type(8)));
typedef float  v4f   __attribute__((ext_vector_type(4)));
typedef int    v4i   __attribute__((ext_vector_type(4)));

union FragU { v16bf bf; v4f f4[2]; };

__device__ __forceinline__ float sigmoidf_(float x) {
    return 1.0f / (1.0f + __expf(-x));
}

// ---------------- async-to-LDS support (gfx1250) ----------------
#if __has_builtin(__builtin_amdgcn_global_load_async_to_lds_b128)
#define HAS_ASYNC_LDS 1
typedef __attribute__((address_space(1))) char as1_char;
typedef __attribute__((address_space(3))) char as3_char;
typedef __attribute__((address_space(1))) v4i  as1_v4i;
typedef __attribute__((address_space(3))) v4i  as3_v4i;
__device__ __forceinline__ void async_copy16(const void* g, void* l) {
    __builtin_amdgcn_global_load_async_to_lds_b128(
        (as1_v4i*)(as1_char*)(const_cast<void*>(g)),
        (as3_v4i*)(as3_char*)(l), 0, 0);
}
__device__ __forceinline__ void async_wait0() {
#if __has_builtin(__builtin_amdgcn_s_wait_asynccnt)
    __builtin_amdgcn_s_wait_asynccnt(0);
#else
    asm volatile("s_wait_asynccnt 0x0" ::: "memory");
#endif
}
#else
#define HAS_ASYNC_LDS 0
#endif

// ---------------- workspace layout (bytes) ----------------
constexpr size_t OFF_XBF  = 0;                                   // [B*T][KX] bf16
constexpr size_t SZ_XBF   = (size_t)BATCH * SEQ * KX * 2;        // 10.5 MB
constexpr size_t OFF_W0XT = OFF_XBF  + SZ_XBF;                   // [1024][320] bf16
constexpr size_t SZ_W0XT  = (size_t)G4 * KX * 2;
constexpr size_t OFF_W0HT = OFF_W0XT + SZ_W0XT;                  // [1024][256] bf16
constexpr size_t SZ_WHT   = (size_t)G4 * HID * 2;
constexpr size_t OFF_W1XT = OFF_W0HT + SZ_WHT;
constexpr size_t OFF_W1HT = OFF_W1XT + SZ_WHT;
constexpr size_t OFF_H1   = OFF_W1HT + SZ_WHT;                   // [B*T][256] bf16
constexpr size_t SZ_H1    = (size_t)BATCH * SEQ * HID * 2;
constexpr size_t OFF_G    = OFF_H1   + SZ_H1;                    // [T][B][1024] f32 (reused for both layers)

constexpr int SMEM_BYTES = BATCH * G4 * 4 + BATCH * HID * 2;     // 256KB gates + 32KB h = 288KB of 320KB LDS

// =====================================================================
// Kernel 1: pack/transpose weights to bf16, N-major (B fragments then load
// exactly like A fragments: two aligned 16B reads per lane per K-tile).
// =====================================================================
__global__ __launch_bounds__(256) void pack_weights_kernel(
    const float* __restrict__ W0, const float* __restrict__ W1,
    __bf16* __restrict__ w0xT, __bf16* __restrict__ w0hT,
    __bf16* __restrict__ w1xT, __bf16* __restrict__ w1hT)
{
    int idx = blockIdx.x * blockDim.x + threadIdx.x;
    const int N0 = G4 * KX;       // 327680
    const int N1 = G4 * HID;      // 262144
    if (idx < N0) {
        int n = idx / KX, k = idx % KX;
        w0xT[idx] = (k < EMB) ? (__bf16)W0[(size_t)k * G4 + n] : (__bf16)0.0f;
    } else if (idx < N0 + N1) {
        int i = idx - N0; int n = i / HID, k = i % HID;
        w0hT[i] = (__bf16)W0[(size_t)(EMB + k) * G4 + n];
    } else if (idx < N0 + 2 * N1) {
        int i = idx - N0 - N1; int n = i / HID, k = i % HID;
        w1xT[i] = (__bf16)W1[(size_t)k * G4 + n];
    } else if (idx < N0 + 3 * N1) {
        int i = idx - N0 - 2 * N1; int n = i / HID, k = i % HID;
        w1hT[i] = (__bf16)W1[(size_t)(HID + k) * G4 + n];
    }
}

// =====================================================================
// Kernel 2: embedding gather -> bf16, K padded 300->320, rows r = b*T + t.
// =====================================================================
__global__ __launch_bounds__(256) void embed_gather_kernel(
    const long long* __restrict__ x_in, const float* __restrict__ emb,
    __bf16* __restrict__ Xbf)
{
    int idx = blockIdx.x * blockDim.x + threadIdx.x;
    const int TOT = BATCH * SEQ * KX;
    if (idx >= TOT) return;
    int r = idx / KX, k = idx % KX;
    int b = r >> 8, t = r & (SEQ - 1);
    long long tok = x_in[b * SEQ + t];
    Xbf[idx] = (k < EMB) ? (__bf16)emb[(size_t)tok * EMB + k] : (__bf16)0.0f;
}

// =====================================================================
// Kernel 3: big parallel GEMM  G[t][b][n] = sum_k A[r][k]*WT[n][k] + bias[n]
// 2x2 tile-block per wave: 1 fragment load per WMMA (2x reuse on A and B).
// =====================================================================
__global__ __launch_bounds__(256) void gemm_wmma_kernel(
    const __bf16* __restrict__ A, int K,
    const __bf16* __restrict__ WT, const float* __restrict__ bias,
    float* __restrict__ Gout)
{
    int wid  = (blockIdx.x * 256 + threadIdx.x) >> 5;   // 16384 wave-blocks
    int lane = threadIdx.x & 31;
    int nblk = wid & 31;            // 32 N-blocks of 2 tiles (64 N-tiles)
    int mblk = wid >> 5;            // 512 M-blocks of 2 tiles (1024 M-tiles)
    int lrow  = lane & 15;
    int hisel = (lane >> 4) << 3;   // lanes 16..31: +8 rows / +8 K (ISA layout)
    int m0 = mblk * 32 + lrow;
    int n0 = nblk * 32 + lrow;

    const __bf16* Ar0 = A  + (size_t)m0 * K;
    const __bf16* Ar1 = Ar0 + (size_t)16 * K;
    const __bf16* Br0 = WT + (size_t)n0 * K;
    const __bf16* Br1 = Br0 + (size_t)16 * K;

    v8f acc00 = {}, acc01 = {}, acc10 = {}, acc11 = {};
    for (int kt = 0; kt < K; kt += 32) {
        int ko = kt + hisel;
        FragU a0, a1, b0, b1;
        a0.f4[0] = *(const v4f*)(Ar0 + ko);      a0.f4[1] = *(const v4f*)(Ar0 + ko + 16);
        a1.f4[0] = *(const v4f*)(Ar1 + ko);      a1.f4[1] = *(const v4f*)(Ar1 + ko + 16);
        b0.f4[0] = *(const v4f*)(Br0 + ko);      b0.f4[1] = *(const v4f*)(Br0 + ko + 16);
        b1.f4[0] = *(const v4f*)(Br1 + ko);      b1.f4[1] = *(const v4f*)(Br1 + ko + 16);
        acc00 = __builtin_amdgcn_wmma_f32_16x16x32_bf16(false, a0.bf, false, b0.bf, (short)0, acc00, false, false);
        acc01 = __builtin_amdgcn_wmma_f32_16x16x32_bf16(false, a0.bf, false, b1.bf, (short)0, acc01, false, false);
        acc10 = __builtin_amdgcn_wmma_f32_16x16x32_bf16(false, a1.bf, false, b0.bf, (short)0, acc10, false, false);
        acc11 = __builtin_amdgcn_wmma_f32_16x16x32_bf16(false, a1.bf, false, b1.bf, (short)0, acc11, false, false);
    }
    #pragma unroll
    for (int mt = 0; mt < 2; ++mt) {
        int rbase = mblk * 32 + mt * 16 + hisel;
        #pragma unroll
        for (int nt = 0; nt < 2; ++nt) {
            int ncol = nblk * 32 + nt * 16 + lrow;
            float bv = bias[ncol];
            const v8f& acc = mt ? (nt ? acc11 : acc10) : (nt ? acc01 : acc00);
            #pragma unroll
            for (int j = 0; j < 8; ++j) {
                int r = rbase + j;
                int b = r >> 8, t = r & (SEQ - 1);
                Gout[(((size_t)t * BATCH + b) << 10) + ncol] = acc[j] + bv;
            }
        }
    }
}

// =====================================================================
// Kernel 4: persistent LSTM recurrence. 1 workgroup, 1024 threads (32 waves).
// LDS: gates f32 [64][1024] (256KB, doubles as async landing zone for G[t])
//      + h bf16 [64][256] (32KB). Cell state c in registers.
// Per step: D = h @ WhT (WMMA, acc from 0), overlapped with an async
// GLOBAL_LOAD_ASYNC_TO_LDS copy of G[t] issued during the previous step;
// then gates = D + G[t] (LDS), then elementwise gate math.
// =====================================================================
__global__ __launch_bounds__(1024) void lstm_recurrent_kernel(
    const float* __restrict__ Gin,         // [T][64][1024] f32 (bias already added)
    const __bf16* __restrict__ WhT,        // [1024][256] bf16, N-major
    __bf16* __restrict__ h_seq_out,        // [B*T][256] bf16 or nullptr
    float* __restrict__ final_out)         // [64][256] f32 or nullptr
{
    extern __shared__ char smem[];
    float*  gates = (float*)smem;                          // [64][1024]
    __bf16* hbuf  = (__bf16*)(smem + BATCH * G4 * 4);      // [64][256]

    int tid  = threadIdx.x;
    int lane = tid & 31;
    int wave = tid >> 5;
    int lrow  = lane & 15;
    int hisel = (lane >> 4) << 3;
    int eb    = tid >> 4;          // elementwise: batch row
    int ebase = (tid & 15) << 4;   // elementwise: 16 hidden units per thread

    for (int i = tid; i < BATCH * HID; i += 1024) hbuf[i] = (__bf16)0.0f;
    float c_reg[16];
    #pragma unroll
    for (int s = 0; s < 16; ++s) c_reg[s] = 0.0f;

#if HAS_ASYNC_LDS
    {   // preload G[0] into the gates LDS buffer (each thread copies 256B)
        const char* gsrc = (const char*)Gin + (size_t)tid * 256;
        char*       ldst = (char*)gates + (size_t)tid * 256;
        #pragma unroll
        for (int i = 0; i < 16; ++i)
            async_copy16(gsrc + i * 16, ldst + i * 16);
    }
#endif
    __syncthreads();

    for (int t = 0; t < SEQ; ++t) {
        const float* Gt = Gin + ((size_t)t << 16);   // t*64*1024

        // ---- phase 1: D = h @ WhT ; 32 waves x (2 blocks of 2x2 tiles) ----
        #pragma unroll 1
        for (int ib = 0; ib < 2; ++ib) {
            int blk  = wave * 2 + ib;                // 64 blocks: 2 Mblk x 32 Nblk
            int mblk = blk >> 5, nblk = blk & 31;
            int m0 = mblk * 32 + lrow;               // batch rows
            int n0 = nblk * 32 + lrow;               // gate columns

            const __bf16* Ah0 = hbuf + m0 * HID;
            const __bf16* Ah1 = Ah0 + 16 * HID;
            const __bf16* Bw0 = WhT + (size_t)n0 * HID;
            const __bf16* Bw1 = Bw0 + (size_t)16 * HID;

            v8f acc00 = {}, acc01 = {}, acc10 = {}, acc11 = {};
            #pragma unroll
            for (int kt = 0; kt < HID; kt += 32) {
                int ko = kt + hisel;
                FragU a0, a1, b0, b1;
                a0.f4[0] = *(const v4f*)(Ah0 + ko);  a0.f4[1] = *(const v4f*)(Ah0 + ko + 16); // ds_read_b128
                a1.f4[0] = *(const v4f*)(Ah1 + ko);  a1.f4[1] = *(const v4f*)(Ah1 + ko + 16);
                b0.f4[0] = *(const v4f*)(Bw0 + ko);  b0.f4[1] = *(const v4f*)(Bw0 + ko + 16); // L2-hot weights
                b1.f4[0] = *(const v4f*)(Bw1 + ko);  b1.f4[1] = *(const v4f*)(Bw1 + ko + 16);
                acc00 = __builtin_amdgcn_wmma_f32_16x16x32_bf16(false, a0.bf, false, b0.bf, (short)0, acc00, false, false);
                acc01 = __builtin_amdgcn_wmma_f32_16x16x32_bf16(false, a0.bf, false, b1.bf, (short)0, acc01, false, false);
                acc10 = __builtin_amdgcn_wmma_f32_16x16x32_bf16(false, a1.bf, false, b0.bf, (short)0, acc10, false, false);
                acc11 = __builtin_amdgcn_wmma_f32_16x16x32_bf16(false, a1.bf, false, b1.bf, (short)0, acc11, false, false);
            }

            // fold in G[t] and publish gates
            if (ib == 0) {
#if HAS_ASYNC_LDS
                async_wait0();       // our async copies of G[t] done
#endif
                __syncthreads();     // ...and everyone else's
            }
            #pragma unroll
            for (int mt = 0; mt < 2; ++mt) {
                int rbase = mblk * 32 + mt * 16 + hisel;
                #pragma unroll
                for (int nt = 0; nt < 2; ++nt) {
                    int ncol = nblk * 32 + nt * 16 + lrow;
                    const v8f& acc = mt ? (nt ? acc11 : acc10) : (nt ? acc01 : acc00);
                    #pragma unroll
                    for (int j = 0; j < 8; ++j) {
                        size_t gidx = ((size_t)(rbase + j) << 10) + ncol;
#if HAS_ASYNC_LDS
                        gates[gidx] += acc[j];                 // G[t] already in LDS
#else
                        gates[gidx] = acc[j] + Gt[gidx];       // direct global fetch
#endif
                    }
                }
            }
        }
        __syncthreads();

        // ---- phase 2: elementwise gate math (gate order i, j, f, o) ----
#if !HAS_ASYNC_LDS
        if (t + 1 < SEQ)
            __builtin_prefetch(Gin + (((size_t)(t + 1)) << 16) + tid * 64, 0, 1);
#endif
        const float* grow = gates + ((size_t)eb << 10);
        #pragma unroll
        for (int s = 0; s < 16; ++s) {
            int hh = ebase + s;
            float gi = grow[hh];
            float gj = grow[HID + hh];
            float gf = grow[2 * HID + hh];
            float go = grow[3 * HID + hh];
            float c = c_reg[s] * sigmoidf_(gf + FORGET_BIAS)
                    + sigmoidf_(gi) * tanhf(gj);
            float h = tanhf(c) * sigmoidf_(go);
            c_reg[s] = c;
            hbuf[eb * HID + hh] = (__bf16)h;
            if (h_seq_out)
                h_seq_out[(size_t)(eb * SEQ + t) * HID + hh] = (__bf16)h;
            if (final_out && t == SEQ - 1)
                final_out[eb * HID + hh] = h;
        }
        __syncthreads();   // gates fully consumed; hbuf published

#if HAS_ASYNC_LDS
        if (t + 1 < SEQ) {  // overlap next step's G fetch with next WMMA phase
            const char* gsrc = (const char*)(Gin + ((size_t)(t + 1) << 16)) + (size_t)tid * 256;
            char*       ldst = (char*)gates + (size_t)tid * 256;
            #pragma unroll
            for (int i = 0; i < 16; ++i)
                async_copy16(gsrc + i * 16, ldst + i * 16);
        }
#endif
    }
}

// =====================================================================
// Host orchestration
// =====================================================================
extern "C" void kernel_launch(void* const* d_in, const int* in_sizes, int n_in,
                              void* d_out, int out_size, void* d_ws, size_t ws_size,
                              hipStream_t stream) {
    const long long* x_in      = (const long long*)d_in[0];
    const float*     embedding = (const float*)d_in[1];
    const float*     W0        = (const float*)d_in[2];
    const float*     b0        = (const float*)d_in[3];
    const float*     W1        = (const float*)d_in[4];
    const float*     b1        = (const float*)d_in[5];
    float* out = (float*)d_out;

    char* ws = (char*)d_ws;
    __bf16* Xbf   = (__bf16*)(ws + OFF_XBF);
    __bf16* W0xT  = (__bf16*)(ws + OFF_W0XT);
    __bf16* W0hT  = (__bf16*)(ws + OFF_W0HT);
    __bf16* W1xT  = (__bf16*)(ws + OFF_W1XT);
    __bf16* W1hT  = (__bf16*)(ws + OFF_W1HT);
    __bf16* H1bf  = (__bf16*)(ws + OFF_H1);
    float*  Gbuf  = (float*)(ws + OFF_G);

    (void)hipFuncSetAttribute((const void*)lstm_recurrent_kernel,
                              hipFuncAttributeMaxDynamicSharedMemorySize, SMEM_BYTES);

    {   // 1) pack weights (bf16, N-major)
        int total = G4 * KX + 3 * G4 * HID;
        pack_weights_kernel<<<(total + 255) / 256, 256, 0, stream>>>(
            W0, W1, W0xT, W0hT, W1xT, W1hT);
    }
    {   // 2) embedding gather -> bf16 padded
        int total = BATCH * SEQ * KX;
        embed_gather_kernel<<<(total + 255) / 256, 256, 0, stream>>>(
            x_in, embedding, Xbf);
    }
    // 3) G0 = X @ W0x + b0  (all timesteps at once)
    gemm_wmma_kernel<<<2048, 256, 0, stream>>>(Xbf, KX, W0xT, b0, Gbuf);
    // 4) layer-0 recurrence (persistent), emits h1 for all t as bf16
    lstm_recurrent_kernel<<<1, 1024, SMEM_BYTES, stream>>>(
        Gbuf, W0hT, H1bf, nullptr);
    // 5) G1 = h1 @ W1x + b1
    gemm_wmma_kernel<<<2048, 256, 0, stream>>>(H1bf, HID, W1xT, b1, Gbuf);
    // 6) layer-1 recurrence, writes final h2[:, -1, :]
    lstm_recurrent_kernel<<<1, 1024, SMEM_BYTES, stream>>>(
        Gbuf, W1hT, nullptr, out);
}
